// IDMRFLoss_51299089383643
// MI455X (gfx1250) — matrix operand order repaired and checked by reading, not compiled, not run on hardware
//
#include <hip/hip_runtime.h>
#include <hip/hip_bf16.h>

typedef _Float16 half_t;
typedef __attribute__((ext_vector_type(16))) _Float16 v16h;
typedef __attribute__((ext_vector_type(8)))  float    v8f;
typedef __attribute__((ext_vector_type(4)))  float    f32x4;
typedef __attribute__((ext_vector_type(4)))  int      v4i;

#define TM 128          // rows per block (8 waves x 16)
#define TN 64           // cols per block
#define BK 32           // K per WMMA step
#define BSTRIDE 40      // halfs per staged column (32 + pad) -> 80B, 16B aligned
#define NEG_BIG -3.0e38f

#if __has_builtin(__builtin_amdgcn_global_load_async_to_lds_b128)
#define HAVE_ASYNC_LDS 1
#else
#define HAVE_ASYNC_LDS 0
#endif

// ---------------------------------------------------------------------------
// async global->LDS staging helpers (CDNA5 GLOBAL_LOAD_ASYNC_TO_LDS_B128,
// tracked by ASYNCcnt); guarded fallback keeps compilation safe.
// Builtin signature (from hipcc diagnostic): params are v4i pointers in
// global (AS1) / LDS (AS3) address spaces.
// ---------------------------------------------------------------------------
#if HAVE_ASYNC_LDS
typedef __attribute__((address_space(1))) v4i  gv4i_t;
typedef __attribute__((address_space(3))) v4i  lv4i_t;
typedef __attribute__((address_space(1))) void gvoid_t;
typedef __attribute__((address_space(3))) void lvoid_t;

__device__ __forceinline__ void async_copy_b128(const half_t* g, half_t* l) {
    __builtin_amdgcn_global_load_async_to_lds_b128(
        (gv4i_t*)(gvoid_t*)(g),
        (lv4i_t*)(lvoid_t*)(l),
        0, 0);
}
__device__ __forceinline__ void wait_asynccnt0() {
#if __has_builtin(__builtin_amdgcn_s_wait_asynccnt)
    __builtin_amdgcn_s_wait_asynccnt(0);
#else
    asm volatile("s_wait_asynccnt 0x0" ::: "memory");
#endif
}
#endif

// ---------------------------------------------------------------------------
// sign-aware float atomic max (works for +/- values, init to NEG_BIG)
// ---------------------------------------------------------------------------
__device__ __forceinline__ void atomicMaxF(float* addr, float value) {
    if (value >= 0.0f) {
        atomicMax((int*)addr, __float_as_int(value));
    } else {
        atomicMin((unsigned int*)addr, __float_as_uint(value));
    }
}

// ---------------------------------------------------------------------------
// A-fragment loader matching CDNA5 WMMA 16-bit A 16x32 VGPR layout (wave32):
// lanes 0-15 -> M=lane, K {0..7} then {16..23};
// lanes 16-31 -> M=lane-16, K {8..15} then {24..31}.
// tn is row-major [P][C] f16, rows 16B-aligned (C=256/512).
// ---------------------------------------------------------------------------
__device__ __forceinline__ v16h load_a_frag(const half_t* __restrict__ base,
                                            int C, int rowBase, int k0, int lane) {
    const int m   = lane & 15;
    const int hlf = lane >> 4;
    const half_t* p = base + (size_t)(rowBase + m) * C + k0 + hlf * 8;
    union { f32x4 f[2]; v16h h; } u;
    u.f[0] = *(const f32x4*)(p);        // K = k0 + hlf*8   .. +7
    u.f[1] = *(const f32x4*)(p + 16);   // K = k0+16+hlf*8  .. +7
    return u.h;
}

// ---------------------------------------------------------------------------
// prep: per spatial site p, compute channel-mean of target, subtract from
// gen/target, L2-normalize over channels, store f16 row-major [B][P][C].
// Inputs are [B][C][H*W] f32 (coalesced over p).
// ---------------------------------------------------------------------------
__global__ __launch_bounds__(256) void prep_kernel(
        const float* __restrict__ gen, const float* __restrict__ tar,
        half_t* __restrict__ gn, half_t* __restrict__ tn, int C, int P) {
    const int b = blockIdx.y;
    const int p = blockIdx.x * blockDim.x + threadIdx.x;  // P % 256 == 0
    const float* g = gen + (size_t)b * C * P;
    const float* t = tar + (size_t)b * C * P;

    float s = 0.0f;
    for (int c = 0; c < C; ++c) s += t[(size_t)c * P + p];
    const float mean = s / (float)C;

    float sg = 0.0f, st = 0.0f;
    for (int c = 0; c < C; ++c) {
        float gf = g[(size_t)c * P + p] - mean;
        float tf = t[(size_t)c * P + p] - mean;
        sg += gf * gf;
        st += tf * tf;
    }
    const float ig = rsqrtf(sg);
    const float it = rsqrtf(st);

    half_t* go = gn + ((size_t)b * P + p) * C;
    half_t* to = tn + ((size_t)b * P + p) * C;
    for (int c = 0; c < C; ++c) {
        go[c] = (half_t)((g[(size_t)c * P + p] - mean) * ig);
        to[c] = (half_t)((t[(size_t)c * P + p] - mean) * it);
    }
}

// ---------------------------------------------------------------------------
// Fused Gram-matrix kernel, recomputed 3x with different epilogues:
//  mode 0: colmax[q] = max_p sim[p][q]
//  mode 1: colsum[q] = sum_p exp(2*(1 - cd/(cdmin+eps)))
//  mode 2: rowmax[p] = max_q dist[p][q]/colsum[q]
// Block: 256 threads = 8 waves; macro-tile 128x64; wave w owns rows w*16..+15.
// B tile (64 cols x 32 K) is staged into LDS (async DMA when available),
// double buffered, shared by all 8 waves; the next A fragment and next B
// tile are in flight while the current step's WMMAs execute.
// ---------------------------------------------------------------------------
__global__ __launch_bounds__(256) void mrf_gemm(
        const half_t* __restrict__ tn, const half_t* __restrict__ gnrm,
        float* __restrict__ colmax, float* __restrict__ colsum,
        float* __restrict__ rowmax, int C, int P, int mode) {
    const int b = blockIdx.z;
    tn   += (size_t)b * P * C;
    gnrm += (size_t)b * P * C;
    colmax += (size_t)b * P;
    colsum += (size_t)b * P;
    rowmax += (size_t)b * P;

    const int lane = threadIdx.x & 31;
    const int wave = threadIdx.x >> 5;
    const int rowBase = blockIdx.y * TM + wave * 16;
    const int colBase = blockIdx.x * TN;

    __shared__ half_t bsh[2][TN * BSTRIDE];   // 2 x 5120 B

    // cooperative B staging: thread -> (col, 8-half chunk)
    const int scol = threadIdx.x >> 2;        // 0..63
    const int skq  = threadIdx.x & 3;         // 0..3 (8 halfs each)
    const half_t* bsrc = gnrm + (size_t)(colBase + scol) * C + skq * 8;
    half_t* const sdst0 = &bsh[0][scol * BSTRIDE + skq * 8];
    half_t* const sdst1 = &bsh[1][scol * BSTRIDE + skq * 8];

    const int nsteps = C >> 5;

    v8f acc[4] = {};

#if HAVE_ASYNC_LDS
    // prologue: async-stage step 0 directly into LDS
    async_copy_b128(bsrc, sdst0);
    v16h a_cur = load_a_frag(tn, C, rowBase, 0, lane);
    wait_asynccnt0();
    __syncthreads();
#else
    // prologue: stage step 0 via registers, prefetch step 1
    f32x4 breg = *(const f32x4*)(bsrc);
    *(f32x4*)sdst0 = breg;
    if (nsteps > 1) breg = *(const f32x4*)(bsrc + BK);
    v16h a_cur = load_a_frag(tn, C, rowBase, 0, lane);
    __syncthreads();
#endif

    const int n   = lane & 15;
    const int hlf = lane >> 4;

    for (int step = 0; step < nsteps; ++step) {
        const int s = step & 1;
#if HAVE_ASYNC_LDS
        // launch async DMA of step+1's B tile into the other buffer
        if (step + 1 < nsteps)
            async_copy_b128(bsrc + (size_t)(step + 1) * BK, s ? sdst0 : sdst1);
#else
        // store step+1's B tile into the other buffer; prefetch step+2
        if (step + 1 < nsteps) {
            *(f32x4*)(s ? sdst0 : sdst1) = breg;
            if (step + 2 < nsteps) breg = *(const f32x4*)(bsrc + (size_t)(step + 2) * BK);
        }
#endif
        v16h a_use = a_cur;
        if (step + 1 < nsteps) a_cur = load_a_frag(tn, C, rowBase, (step + 1) * BK, lane);

#pragma unroll
        for (int nsub = 0; nsub < 4; ++nsub) {
            // B 32x16 fragment: lane -> col n, K = hlf*8 .. +7 then 16+hlf*8 .. +7
            const half_t* p = &bsh[s][(nsub * 16 + n) * BSTRIDE + hlf * 8];
            union { f32x4 f[2]; v16h h; } u;
            u.f[0] = *(const f32x4*)(p);
            u.f[1] = *(const f32x4*)(p + 16);
            acc[nsub] = __builtin_amdgcn_wmma_f32_16x16x32_f16(
                false, a_use, false, u.h, (short)0, acc[nsub], false, false);
        }
#if HAVE_ASYNC_LDS
        wait_asynccnt0();   // our slice of step+1's tile is in LDS
#endif
        __syncthreads();
    }

    // D layout: VGPR r, lane l -> row = rowBase + r + 8*(l>>4), col = colBase+16*nsub+(l&15)
    __shared__ float sred[TN];

    if (mode == 0) {
        if (threadIdx.x < TN) sred[threadIdx.x] = NEG_BIG;
        __syncthreads();
#pragma unroll
        for (int nsub = 0; nsub < 4; ++nsub) {
            float m = acc[nsub][0];
#pragma unroll
            for (int r = 1; r < 8; ++r) m = fmaxf(m, acc[nsub][r]);
            m = fmaxf(m, __shfl_xor(m, 16));     // combine the two row halves
            if (hlf == 0) atomicMaxF(&sred[nsub * 16 + n], m);
        }
        __syncthreads();
        if (threadIdx.x < TN) atomicMaxF(&colmax[colBase + threadIdx.x], sred[threadIdx.x]);
    } else if (mode == 1) {
        if (threadIdx.x < TN) sred[threadIdx.x] = 0.0f;
        __syncthreads();
#pragma unroll
        for (int nsub = 0; nsub < 4; ++nsub) {
            const int col = colBase + nsub * 16 + n;
            const float denom = (1.0f - colmax[col]) * 0.5f + 1e-5f;  // cdmin + eps
            const float invd  = 1.0f / denom;
            float s = 0.0f;
#pragma unroll
            for (int r = 0; r < 8; ++r) {
                float cd = (1.0f - acc[nsub][r]) * 0.5f;
                s += __expf(2.0f * (1.0f - cd * invd));   // (BIAS-rel)/SIGMA
            }
            s += __shfl_xor(s, 16);
            if (hlf == 0) atomicAdd(&sred[nsub * 16 + n], s);
        }
        __syncthreads();
        if (threadIdx.x < TN) atomicAdd(&colsum[colBase + threadIdx.x], sred[threadIdx.x]);
    } else {
        float v[8];
#pragma unroll
        for (int r = 0; r < 8; ++r) v[r] = NEG_BIG;
#pragma unroll
        for (int nsub = 0; nsub < 4; ++nsub) {
            const int col = colBase + nsub * 16 + n;
            const float denom = (1.0f - colmax[col]) * 0.5f + 1e-5f;
            const float invd  = 1.0f / denom;
            const float invs  = 1.0f / colsum[col];
#pragma unroll
            for (int r = 0; r < 8; ++r) {
                float cd = (1.0f - acc[nsub][r]) * 0.5f;
                float d  = __expf(2.0f * (1.0f - cd * invd)) * invs;
                v[r] = fmaxf(v[r], d);
            }
        }
        // butterfly over the 16 lanes sharing one row half
#pragma unroll
        for (int r = 0; r < 8; ++r) {
#pragma unroll
            for (int m = 1; m < 16; m <<= 1) v[r] = fmaxf(v[r], __shfl_xor(v[r], m));
        }
#pragma unroll
        for (int r = 0; r < 8; ++r)
            if (n == r) atomicMaxF(&rowmax[rowBase + r + 8 * hlf], v[r]);
    }
}

// ---------------------------------------------------------------------------
// init stats [colmax3|colsum3|rowmax3|colmax4|colsum4|rowmax4] and zero out
// ---------------------------------------------------------------------------
__global__ void init_kernel(float* __restrict__ stats, float* __restrict__ out,
                            int N3, int N4) {
    const int i = blockIdx.x * blockDim.x + threadIdx.x;
    const int total = 3 * N3 + 3 * N4;
    if (i < total) {
        bool is_sum = (i >= N3 && i < 2 * N3) ||
                      (i >= 3 * N3 + N4 && i < 3 * N3 + 2 * N4);
        stats[i] = is_sum ? 0.0f : NEG_BIG;
    }
    if (i == 0) *out = 0.0f;
}

// ---------------------------------------------------------------------------
// finalize: out += -w * log(mean_p rowmax[b][p]) per batch
// ---------------------------------------------------------------------------
__global__ __launch_bounds__(256) void finalize_kernel(
        const float* __restrict__ rowmax, int P, float w, float* __restrict__ out) {
    const int b = blockIdx.x;
    const float* rm = rowmax + (size_t)b * P;
    float s = 0.0f;
    for (int p = threadIdx.x; p < P; p += blockDim.x) s += rm[p];
    __shared__ float sm[256];
    sm[threadIdx.x] = s;
    __syncthreads();
    for (int off = 128; off > 0; off >>= 1) {
        if (threadIdx.x < off) sm[threadIdx.x] += sm[threadIdx.x + off];
        __syncthreads();
    }
    if (threadIdx.x == 0) atomicAdd(out, -w * logf(sm[0] / (float)P));
}

// ---------------------------------------------------------------------------
extern "C" void kernel_launch(void* const* d_in, const int* in_sizes, int n_in,
                              void* d_out, int out_size, void* d_ws, size_t ws_size,
                              hipStream_t stream) {
    const float* gen3 = (const float*)d_in[0];
    const float* tar3 = (const float*)d_in[1];
    const float* gen4 = (const float*)d_in[2];
    const float* tar4 = (const float*)d_in[3];
    float* out = (float*)d_out;

    const int B = 4;
    const int C3 = 256, P3 = 64 * 64;   // relu3_2
    const int C4 = 512, P4 = 32 * 32;   // relu4_2

    // workspace layout: f16 features then f32 stats
    half_t* gn3 = (half_t*)d_ws;
    half_t* tn3 = gn3 + (size_t)B * P3 * C3;
    half_t* gn4 = tn3 + (size_t)B * P3 * C3;
    half_t* tn4 = gn4 + (size_t)B * P4 * C4;
    float* stats   = (float*)(tn4 + (size_t)B * P4 * C4);
    float* colmax3 = stats;
    float* colsum3 = colmax3 + B * P3;
    float* rowmax3 = colsum3 + B * P3;
    float* colmax4 = rowmax3 + B * P3;
    float* colsum4 = colmax4 + B * P4;
    float* rowmax4 = colsum4 + B * P4;

    const int nstat = 3 * B * P3 + 3 * B * P4;
    init_kernel<<<(nstat + 255) / 256, 256, 0, stream>>>(stats, out, B * P3, B * P4);

    prep_kernel<<<dim3(P3 / 256, B), 256, 0, stream>>>(gen3, tar3, gn3, tn3, C3, P3);
    prep_kernel<<<dim3(P4 / 256, B), 256, 0, stream>>>(gen4, tar4, gn4, tn4, C4, P4);

    const dim3 g3(P3 / TN, P3 / TM, B);
    const dim3 g4(P4 / TN, P4 / TM, B);
    for (int mode = 0; mode < 3; ++mode) {
        mrf_gemm<<<g3, 256, 0, stream>>>(tn3, gn3, colmax3, colsum3, rowmax3, C3, P3, mode);
        mrf_gemm<<<g4, 256, 0, stream>>>(tn4, gn4, colmax4, colsum4, rowmax4, C4, P4, mode);
    }

    finalize_kernel<<<B, 256, 0, stream>>>(rowmax3, P3, 1.0f, out);
    finalize_kernel<<<B, 256, 0, stream>>>(rowmax4, P4, 2.0f, out);  // style + content
}
